// HierarchicalRNN_46926812676406
// MI455X (gfx1250) — matrix operand order repaired and checked by reading, not compiled, not run on hardware
//
#include <hip/hip_runtime.h>
#include <hip/hip_bf16.h>
#include <math.h>

// ---------------- architecture constants (match reference) ----------------
#define A_NUM   4
#define E_PER   512
#define I_PER   128
#define H_DIM   2560              // 4*(512+128)
#define E_TOT   2048              // A_NUM*E_PER
#define IN_DIM  128
#define KC      (H_DIM + IN_DIM)  // 2688 concatenated K
#define B_DIM   64
#define T_DIM   500
#define ALPHA_X 0.2f              // DT/TAU_X

typedef __attribute__((ext_vector_type(16))) __bf16 v16bf;
typedef __attribute__((ext_vector_type(8)))  __bf16 v8bf;
typedef __attribute__((ext_vector_type(8)))  float  v8f;

// ---------------------------------------------------------------------------
// Prep: build concatenated, signed-masked weight panel in K-major bf16 layout
//   Wb[k][n] = (k <  H) ? |W_rec[n,k]| * rec_mask(n,k)
//            : (k >= H) ? |W_in [n,k-H]| * in_mask(n)
// K-major so each k row of a WMMA B-fragment is 16 contiguous bf16 across lanes.
// ---------------------------------------------------------------------------
__device__ __forceinline__ int area_of(int h) {
    return (h < E_TOT) ? (h >> 9) : ((h - E_TOT) >> 7);
}

__global__ void prep_weights(const float* __restrict__ W_rec,
                             const float* __restrict__ W_in,
                             __bf16* __restrict__ Wb) {
    size_t idx = (size_t)blockIdx.x * 256 + threadIdx.x;
    if (idx >= (size_t)KC * H_DIM) return;
    int k = (int)(idx / H_DIM);
    int n = (int)(idx % H_DIM);
    float w;
    if (k < H_DIM) {
        float v = fabsf(W_rec[(size_t)n * H_DIM + k]);
        int ao = area_of(n), ai = area_of(k);
        bool col_exc = (k < E_TOT);
        int d = ao - ai; if (d < 0) d = -d;
        bool conn = col_exc ? (d <= 1) : (ao == ai);   // E cols: tri-diag; I cols: within area
        float sgn = col_exc ? 1.0f : -1.0f;            // Dale's law sign by presynaptic type
        w = (conn && (n != k)) ? v * sgn : 0.0f;       // remove_diag
    } else {
        int i = k - H_DIM;
        float v = fabsf(W_in[(size_t)n * IN_DIM + i]);
        w = (area_of(n) == 0) ? v : 0.0f;              // only area 0 receives input
    }
    Wb[idx] = (__bf16)w;
}

// zero state: x (f32) and r0 (bf16); retanh(0) == 0
__global__ void init_state(float* __restrict__ x, __bf16* __restrict__ r0) {
    int idx = blockIdx.x * 256 + threadIdx.x;
    if (idx >= B_DIM * H_DIM) return;
    x[idx]  = 0.0f;
    r0[idx] = (__bf16)0.0f;
}

// ---------------------------------------------------------------------------
// One RNN step:  pre = [r | u_t] * Wb  (M=64, N=2560, K=2688) via bf16 WMMA,
// then leaky integration + retanh epilogue.
// Block = 4 waves; wave w owns N-tile (blockIdx*64 + w*16) and all 4 M-tiles.
// K reduction is split: branch-free main loop over the recurrent 2560 K,
// then a fully unrolled 4-iteration tail for the folded input projection.
// ---------------------------------------------------------------------------
__global__ __launch_bounds__(128) void rnn_step(
    const __bf16* __restrict__ Wb,     // [KC][H]  K-major weight panel (L2-resident)
    const __bf16* __restrict__ r_in,   // [B][H]   previous rates, bf16
    const float*  __restrict__ u_t,    // [B][IN]  raw inputs for this step
    const float*  __restrict__ b_rec,  // [H]
    float*        __restrict__ x,      // [B][H]   membrane state (in/out)
    __bf16*       __restrict__ r_out,  // [B][H]   new rates, bf16 (ping-pong)
    float*        __restrict__ rates)  // [B][H]   d_out slice for this step
{
    const int wave = threadIdx.x >> 5;
    const int lane = threadIdx.x & 31;
    const int lo   = lane & 15;
    const bool hiH = lane >= 16;
    const int ncol = blockIdx.x * 64 + wave * 16;

    v8f acc0 = {}, acc1 = {}, acc2 = {}, acc3 = {};

    // A fragment K groups per ISA layout:
    //  lanes 0-15: K in {0..7, 16..23};  lanes 16-31: K in {8..15, 24..31}
    const int kOff = hiH ? 8 : 0;

    // Per-M-tile A row pointers (advance by 32 bf16 per iteration)
    const __bf16* a0p = r_in + (size_t)(0 * 16 + lo) * H_DIM + kOff;
    const __bf16* a1p = r_in + (size_t)(1 * 16 + lo) * H_DIM + kOff;
    const __bf16* a2p = r_in + (size_t)(2 * 16 + lo) * H_DIM + kOff;
    const __bf16* a3p = r_in + (size_t)(3 * 16 + lo) * H_DIM + kOff;
    // B pointer: column n = ncol + lo, K base kOff (same table K order as A)
    const __bf16* bp  = Wb + (size_t)kOff * H_DIM + ncol + lo;

#pragma unroll 2
    for (int kb = 0; kb < H_DIM; kb += 32) {
        v16bf bfrag; for (int s = 0; s < 16; ++s) bfrag[s] = bp[(size_t)(s + (s >= 8 ? 8 : 0)) * H_DIM];
        bp += (size_t)32 * H_DIM;

        v8bf a0l = *(const v8bf*)(a0p);      v8bf a0h = *(const v8bf*)(a0p + 16);
        v8bf a1l = *(const v8bf*)(a1p);      v8bf a1h = *(const v8bf*)(a1p + 16);
        v8bf a2l = *(const v8bf*)(a2p);      v8bf a2h = *(const v8bf*)(a2p + 16);
        v8bf a3l = *(const v8bf*)(a3p);      v8bf a3h = *(const v8bf*)(a3p + 16);
        a0p += 32; a1p += 32; a2p += 32; a3p += 32;

        v16bf af0 = __builtin_shufflevector(a0l, a0h, 0,1,2,3,4,5,6,7,8,9,10,11,12,13,14,15);
        v16bf af1 = __builtin_shufflevector(a1l, a1h, 0,1,2,3,4,5,6,7,8,9,10,11,12,13,14,15);
        v16bf af2 = __builtin_shufflevector(a2l, a2h, 0,1,2,3,4,5,6,7,8,9,10,11,12,13,14,15);
        v16bf af3 = __builtin_shufflevector(a3l, a3h, 0,1,2,3,4,5,6,7,8,9,10,11,12,13,14,15);

        acc0 = __builtin_amdgcn_wmma_f32_16x16x32_bf16(false, af0, false, bfrag, (short)0, acc0, false, false);
        acc1 = __builtin_amdgcn_wmma_f32_16x16x32_bf16(false, af1, false, bfrag, (short)0, acc1, false, false);
        acc2 = __builtin_amdgcn_wmma_f32_16x16x32_bf16(false, af2, false, bfrag, (short)0, acc2, false, false);
        acc3 = __builtin_amdgcn_wmma_f32_16x16x32_bf16(false, af3, false, bfrag, (short)0, acc3, false, false);
    }

    // Folded input projection tail: K = 2560..2688 from raw f32 inputs
    const float* u0p = u_t + (size_t)(0 * 16 + lo) * IN_DIM + kOff;
    const float* u1p = u_t + (size_t)(1 * 16 + lo) * IN_DIM + kOff;
    const float* u2p = u_t + (size_t)(2 * 16 + lo) * IN_DIM + kOff;
    const float* u3p = u_t + (size_t)(3 * 16 + lo) * IN_DIM + kOff;

#pragma unroll
    for (int kb = 0; kb < IN_DIM; kb += 32) {
        v16bf bfrag; for (int s = 0; s < 16; ++s) bfrag[s] = bp[(size_t)(s + (s >= 8 ? 8 : 0)) * H_DIM];
        bp += (size_t)32 * H_DIM;

        v8bf a0l = __builtin_convertvector(*(const v8f*)(u0p), v8bf);
        v8bf a0h = __builtin_convertvector(*(const v8f*)(u0p + 16), v8bf);
        v8bf a1l = __builtin_convertvector(*(const v8f*)(u1p), v8bf);
        v8bf a1h = __builtin_convertvector(*(const v8f*)(u1p + 16), v8bf);
        v8bf a2l = __builtin_convertvector(*(const v8f*)(u2p), v8bf);
        v8bf a2h = __builtin_convertvector(*(const v8f*)(u2p + 16), v8bf);
        v8bf a3l = __builtin_convertvector(*(const v8f*)(u3p), v8bf);
        v8bf a3h = __builtin_convertvector(*(const v8f*)(u3p + 16), v8bf);
        u0p += 32; u1p += 32; u2p += 32; u3p += 32;

        v16bf af0 = __builtin_shufflevector(a0l, a0h, 0,1,2,3,4,5,6,7,8,9,10,11,12,13,14,15);
        v16bf af1 = __builtin_shufflevector(a1l, a1h, 0,1,2,3,4,5,6,7,8,9,10,11,12,13,14,15);
        v16bf af2 = __builtin_shufflevector(a2l, a2h, 0,1,2,3,4,5,6,7,8,9,10,11,12,13,14,15);
        v16bf af3 = __builtin_shufflevector(a3l, a3h, 0,1,2,3,4,5,6,7,8,9,10,11,12,13,14,15);

        acc0 = __builtin_amdgcn_wmma_f32_16x16x32_bf16(false, af0, false, bfrag, (short)0, acc0, false, false);
        acc1 = __builtin_amdgcn_wmma_f32_16x16x32_bf16(false, af1, false, bfrag, (short)0, acc1, false, false);
        acc2 = __builtin_amdgcn_wmma_f32_16x16x32_bf16(false, af2, false, bfrag, (short)0, acc2, false, false);
        acc3 = __builtin_amdgcn_wmma_f32_16x16x32_bf16(false, af3, false, bfrag, (short)0, acc3, false, false);
    }

    // Epilogue. C/D layout: lanes 0-15 -> N=lane, VGPR v -> M=v;
    //                       lanes 16-31 -> N=lane-16, VGPR v -> M=v+8.
    const int n = ncol + lo;
    const float bias = b_rec[n];
    const int mBase = hiH ? 8 : 0;
#pragma unroll
    for (int mt = 0; mt < 4; ++mt) {
        v8f acc = (mt == 0) ? acc0 : (mt == 1) ? acc1 : (mt == 2) ? acc2 : acc3;
#pragma unroll
        for (int v = 0; v < 8; ++v) {
            const int m = mt * 16 + v + mBase;
            const size_t idx = (size_t)m * H_DIM + n;
            float pre = acc[v] + bias;
            float xv  = x[idx];
            xv = (1.0f - ALPHA_X) * xv + ALPHA_X * pre;
            x[idx] = xv;
            float r = tanhf(fmaxf(xv, 0.0f));
            r_out[idx] = (__bf16)r;
            __builtin_nontemporal_store(r, rates + idx);  // streamed; keep Wb in L2
        }
    }
}

// ---------------------------------------------------------------------------
extern "C" void kernel_launch(void* const* d_in, const int* in_sizes, int n_in,
                              void* d_out, int out_size, void* d_ws, size_t ws_size,
                              hipStream_t stream) {
    const float* inputs = (const float*)d_in[0];  // [T][B][IN]
    const float* W_rec  = (const float*)d_in[1];  // [H][H]
    const float* b_rec  = (const float*)d_in[2];  // [H]
    const float* W_in   = (const float*)d_in[3];  // [H][IN]
    float* out = (float*)d_out;                   // [T][B][H]

    char* ws = (char*)d_ws;
    __bf16* Wb = (__bf16*)ws;                            // 13.76 MB, K-major
    size_t off = (size_t)KC * H_DIM * sizeof(__bf16);
    float* x = (float*)(ws + off);                       // 0.66 MB membrane state
    off += (size_t)B_DIM * H_DIM * sizeof(float);
    __bf16* r0 = (__bf16*)(ws + off);                    // ping
    off += (size_t)B_DIM * H_DIM * sizeof(__bf16);
    __bf16* r1 = (__bf16*)(ws + off);                    // pong

    {
        size_t nel = (size_t)KC * H_DIM;
        prep_weights<<<(unsigned)((nel + 255) / 256), 256, 0, stream>>>(W_rec, W_in, Wb);
    }
    init_state<<<(B_DIM * H_DIM + 255) / 256, 256, 0, stream>>>(x, r0);

    for (int t = 0; t < T_DIM; ++t) {
        const __bf16* rin  = (t & 1) ? r1 : r0;
        __bf16*       rout = (t & 1) ? r0 : r1;
        rnn_step<<<H_DIM / 64, 128, 0, stream>>>(
            Wb, rin,
            inputs + (size_t)t * B_DIM * IN_DIM,
            b_rec, x, rout,
            out + (size_t)t * B_DIM * H_DIM);
    }
}
